// GTA3Layer_34454227648599
// MI455X (gfx1250) — matrix operand order repaired and checked by reading, not compiled, not run on hardware
//
#include <hip/hip_runtime.h>
#include <hip/hip_bf16.h>

// Problem constants (match reference)
#define BB 2
#define NN 2048
#define DD 128
#define HH 8
#define DH 16

typedef _Float16 v16h __attribute__((ext_vector_type(16)));
typedef _Float16 v8h  __attribute__((ext_vector_type(8)));
typedef float    v8f  __attribute__((ext_vector_type(8)));

static __device__ __forceinline__ v8f wmma_f16(v16h a, v16h b, v8f c) {
    // D = A(16x32 f16) * B(32x16 f16) + C(16x16 f32)
    return __builtin_amdgcn_wmma_f32_16x16x32_f16(
        /*neg_a=*/false, a, /*neg_b=*/false, b,
        /*c_mod=*/(short)0, c, /*reuse_a=*/false, /*reuse_b=*/false);
}

// Load an f16 A-operand fragment (16x32, MxK) from a row-major [16 x ldk] tile.
// ISA layout: lanes 0-15 hold K {0..7, 16..23} of row=lane; lanes 16-31 hold K {8..15, 24..31}.
static __device__ __forceinline__ v16h load_a_f16(const _Float16* __restrict__ p, int ldk,
                                                  int kbase, int lane) {
    const int row = lane & 15;
    const int k0  = kbase + ((lane >> 4) << 3);
    const _Float16* rp = p + (size_t)row * ldk + k0;
    v8h lo = *(const v8h*)(rp);
    v8h hi = *(const v8h*)(rp + 16);
    v16h r;
#pragma unroll
    for (int t = 0; t < 8; ++t) { r[t] = lo[t]; r[8 + t] = hi[t]; }
    return r;
}

// Load an f16 B-operand fragment (32x16, KxN) from packed-transposed weights wt[Nout][K].
// ISA layout: lane holds col=lane%16, K {0..7, 16..23} (lanes 0-15) / {8..15, 24..31} (lanes 16-31).
static __device__ __forceinline__ v16h load_b_packed(const _Float16* __restrict__ wt, int ldk,
                                                     int j, int kbase, int lane) {
    const _Float16* bp = wt + (size_t)j * ldk + kbase + ((lane >> 4) << 3);
    v8h lo = *(const v8h*)(bp);
    v8h hi = *(const v8h*)(bp + 16);
    v16h r;
#pragma unroll
    for (int t = 0; t < 8; ++t) { r[t] = lo[t]; r[8 + t] = hi[t]; }
    return r;
}

// ---------------------------------------------------------------------------
// Prep: transpose fp32 weight [K x Nout] -> f16 packed [Nout x K]
// ---------------------------------------------------------------------------
__global__ __launch_bounds__(256) void k_pack(const float* __restrict__ src,
                                              _Float16* __restrict__ dst,
                                              int K, int Nout) {
    int idx = blockIdx.x * 256 + threadIdx.x;
    if (idx >= K * Nout) return;
    int j = idx / K;
    int k = idx - j * K;
    dst[(size_t)j * K + k] = (_Float16)src[(size_t)k * Nout + j];
}

// ---------------------------------------------------------------------------
// Kernel A: QKV projection. block = 16 tokens x 8 waves; wave w == head w.
// q,k stored f16 [B*H][N][16]; v stored transposed f16 [B*H][16][N].
// ---------------------------------------------------------------------------
__global__ __launch_bounds__(256) void k_qkv(const float* __restrict__ h,
                                             const _Float16* __restrict__ wqt,
                                             const _Float16* __restrict__ wkt,
                                             const _Float16* __restrict__ wvt,
                                             const float* __restrict__ bq,
                                             const float* __restrict__ bk,
                                             const float* __restrict__ bv,
                                             _Float16* __restrict__ qf,
                                             _Float16* __restrict__ kf,
                                             _Float16* __restrict__ vt) {
    const int lane = threadIdx.x & 31;
    const int w    = threadIdx.x >> 5;       // head / col-tile
    const int rowbase = blockIdx.x * 16;     // flat token row over B*N
    const int b    = rowbase >> 11;          // /N
    const int nbase = rowbase & (NN - 1);

    // A-frags from fp32 h (convert), shared across the 3 GEMMs
    const float* hrow = h + (size_t)(rowbase + (lane & 15)) * DD;
    v16h afr[4];
#pragma unroll
    for (int kc = 0; kc < 4; ++kc) {
        int k0 = kc * 32 + ((lane >> 4) << 3);
#pragma unroll
        for (int t = 0; t < 8; ++t) {
            afr[kc][t]     = (_Float16)hrow[k0 + t];
            afr[kc][8 + t] = (_Float16)hrow[k0 + 16 + t];
        }
    }

    const int j = w * 16 + (lane & 15);      // output column (== head*16 + dh)
    v8f aq = {}, ak = {}, av = {};
#pragma unroll
    for (int kc = 0; kc < 4; ++kc) {
        aq = wmma_f16(afr[kc], load_b_packed(wqt, DD, j, kc * 32, lane), aq);
        ak = wmma_f16(afr[kc], load_b_packed(wkt, DD, j, kc * 32, lane), ak);
        av = wmma_f16(afr[kc], load_b_packed(wvt, DD, j, kc * 32, lane), av);
    }

    const float bqv = bq[j], bkv = bk[j], bvv = bv[j];
    const size_t bh = (size_t)b * HH + w;
    const int dh = lane & 15;
#pragma unroll
    for (int r = 0; r < 8; ++r) {
        int rr  = r + ((lane >> 4) << 3);
        int tok = nbase + rr;
        qf[(bh * NN + tok) * DH + dh] = (_Float16)(aq[r] + bqv);
        kf[(bh * NN + tok) * DH + dh] = (_Float16)(ak[r] + bkv);
        vt[(bh * DH + dh) * NN + tok] = (_Float16)(av[r] + bvv);
    }
}

// ---------------------------------------------------------------------------
// Kernel B: flash attention with fused phi reweighting.
// block = (b, 16-row tile), 8 waves = 8 heads. Online softmax over exponent
// arg = s/sqrt(DH) + A*ln(alpha+1e-10), masked -inf where A==0 or col>=len.
// ---------------------------------------------------------------------------
__global__ __launch_bounds__(256) void k_attn(const _Float16* __restrict__ qf,
                                              const _Float16* __restrict__ kf,
                                              const _Float16* __restrict__ vt,
                                              const int* __restrict__ Adj,
                                              const int* __restrict__ lengths,
                                              const float* __restrict__ alphap,
                                              _Float16* __restrict__ attn) {
    __shared__ _Float16 plds[HH][16][32];    // per-wave P transpose staging (C-layout -> A-layout)

    const int lane = threadIdx.x & 31;
    const int head = threadIdx.x >> 5;
    const int b    = blockIdx.x >> 7;        // N/16 = 128 tiles
    const int tile = blockIdx.x & 127;
    const int len  = lengths[b];
    const float la = __logf(fmaxf(alphap[0], 0.0f) + 1e-10f);

    const size_t bh = (size_t)b * HH + head;
    const _Float16* qp = qf + (bh * NN + tile * 16) * DH;
    const _Float16* kp = kf + bh * NN * DH;
    const _Float16* vp = vt + bh * DH * NN;
    const int* arow0 = Adj + ((size_t)b * NN + tile * 16) * NN;

    // Q A-frag, real K = 0..15, K 16..31 zero-padded.
    v16h qa;
    {
        const int row = lane & 15;
        const int k0  = ((lane >> 4) << 3);
        v8h q8 = *(const v8h*)(qp + row * DH + k0);
#pragma unroll
        for (int t = 0; t < 8; ++t) { qa[t] = q8[t]; qa[8 + t] = (_Float16)0.0f; }
    }

    float m[8], l[8];
    v8f oacc = {};
#pragma unroll
    for (int r = 0; r < 8; ++r) { m[r] = -1e30f; l[r] = 0.0f; }

    const int half = lane >> 4;
    const int ll   = lane & 15;

#pragma unroll 1
    for (int c = 0; c < NN; c += 32) {
        // --- scores: two 16x16 sub-tiles (K padded 16->32) ---
        v8f s[2];
#pragma unroll
        for (int sub = 0; sub < 2; ++sub) {
            v16h kb;
            {
                v8h k8 = *(const v8h*)(kp + (size_t)(c + sub * 16 + ll) * DH + (half << 3));
#pragma unroll
                for (int t = 0; t < 8; ++t) { kb[t] = k8[t]; kb[8 + t] = (_Float16)0.0f; }
            }
            v8f z = {};
            s[sub] = wmma_f16(qa, kb, z);
        }

        // --- masked exponent arg = s*0.25 + A*la ---
        float arg[2][8];
#pragma unroll
        for (int sub = 0; sub < 2; ++sub) {
            const int col = c + sub * 16 + ll;
#pragma unroll
            for (int r = 0; r < 8; ++r) {
                const int rr = r + (half << 3);
                int avv = arow0[(size_t)rr * NN + col];
                float sv = s[sub][r] * 0.25f + (float)avv * la;
                arg[sub][r] = (avv != 0 && col < len) ? sv : -1e30f;
            }
        }

        // --- online softmax update, P -> LDS (f16) ---
#pragma unroll
        for (int r = 0; r < 8; ++r) {
            float mx = fmaxf(arg[0][r], arg[1][r]);
#pragma unroll
            for (int off = 8; off >= 1; off >>= 1)
                mx = fmaxf(mx, __shfl_xor(mx, off, 16));
            float mn = fmaxf(m[r], mx);
            float sc = __expf(m[r] - mn);     // both -1e30 -> exp(0)=1, state stays 0
            m[r] = mn;
            float p0 = (arg[0][r] > -1e29f) ? __expf(arg[0][r] - mn) : 0.0f;
            float p1 = (arg[1][r] > -1e29f) ? __expf(arg[1][r] - mn) : 0.0f;
            float ps = p0 + p1;
#pragma unroll
            for (int off = 8; off >= 1; off >>= 1)
                ps += __shfl_xor(ps, off, 16);
            l[r]    = l[r] * sc + ps;
            oacc[r] = oacc[r] * sc;
            const int rr = r + (half << 3);
            plds[head][rr][ll]      = (_Float16)p0;
            plds[head][rr][16 + ll] = (_Float16)p1;
        }

        // --- O += P(16x32) @ V(32x16); DS ops are in-order within a wave ---
        v16h pa;
        {
            const int prow = lane & 15;
            const int pk0  = (lane >> 4) << 3;
#pragma unroll
            for (int t = 0; t < 8; ++t) {
                pa[t]     = plds[head][prow][pk0 + t];
                pa[8 + t] = plds[head][prow][16 + pk0 + t];
            }
        }
        v16h vb;
        {
            const _Float16* vq = vp + (size_t)ll * NN + c + ((lane >> 4) << 3);
            v8h vlo = *(const v8h*)(vq);
            v8h vhi = *(const v8h*)(vq + 16);
#pragma unroll
            for (int t = 0; t < 8; ++t) { vb[t] = vlo[t]; vb[8 + t] = vhi[t]; }
        }
        oacc = wmma_f16(pa, vb, oacc);
    }

    // epilogue: normalize by l, zero padded rows, store f16 [B*N][D]
#pragma unroll
    for (int r = 0; r < 8; ++r) {
        const int rr  = r + (half << 3);
        const int tok = tile * 16 + rr;
        const float inv = (l[r] > 0.0f && tok < len) ? (1.0f / l[r]) : 0.0f;
        attn[((size_t)b * NN + tok) * DD + head * 16 + ll] = (_Float16)(oacc[r] * inv);
    }
}

// ---------------------------------------------------------------------------
// Kernel C: x = LayerNorm(h + attn @ Wo + bo); store x as f32 and f16.
// block = 16 tokens, wave w handles cols [16w, 16w+16).
// ---------------------------------------------------------------------------
__global__ __launch_bounds__(256) void k_proj_ln1(const _Float16* __restrict__ attn,
                                                  const _Float16* __restrict__ wot,
                                                  const float* __restrict__ bo,
                                                  const float* __restrict__ hres,
                                                  const float* __restrict__ g1,
                                                  const float* __restrict__ be1,
                                                  float* __restrict__ x32,
                                                  _Float16* __restrict__ x16) {
    __shared__ float red[16][HH][2];
    const int lane = threadIdx.x & 31;
    const int w    = threadIdx.x >> 5;
    const int rowbase = blockIdx.x * 16;
    const int half = lane >> 4, ll = lane & 15;
    const int j = w * 16 + ll;

    const _Float16* a0 = attn + (size_t)rowbase * DD;
    v8f acc = {};
#pragma unroll
    for (int kc = 0; kc < 4; ++kc)
        acc = wmma_f16(load_a_f16(a0, DD, kc * 32, lane),
                       load_b_packed(wot, DD, j, kc * 32, lane), acc);

    const float bj = bo[j];
    float vals[8];
#pragma unroll
    for (int r = 0; r < 8; ++r) {
        const int grow = rowbase + r + (half << 3);
        vals[r] = acc[r] + bj + hres[(size_t)grow * DD + j];
    }

    // per-row partial sums over this wave's 16 columns
#pragma unroll
    for (int r = 0; r < 8; ++r) {
        float s1 = vals[r], s2 = vals[r] * vals[r];
#pragma unroll
        for (int off = 8; off >= 1; off >>= 1) {
            s1 += __shfl_xor(s1, off, 16);
            s2 += __shfl_xor(s2, off, 16);
        }
        if (ll == 0) {
            const int rr = r + (half << 3);
            red[rr][w][0] = s1;
            red[rr][w][1] = s2;
        }
    }
    __syncthreads();

    const float gj = g1[j], bej = be1[j];
#pragma unroll
    for (int r = 0; r < 8; ++r) {
        const int rr = r + (half << 3);
        float s1 = 0.0f, s2 = 0.0f;
#pragma unroll
        for (int ww = 0; ww < HH; ++ww) { s1 += red[rr][ww][0]; s2 += red[rr][ww][1]; }
        const float mu  = s1 * (1.0f / DD);
        const float var = s2 * (1.0f / DD) - mu * mu;
        const float rs  = rsqrtf(var + 1e-5f);
        const float xn  = (vals[r] - mu) * rs * gj + bej;
        const size_t grow = (size_t)(rowbase + rr);
        x32[grow * DD + j] = xn;
        x16[grow * DD + j] = (_Float16)xn;
    }
}

// ---------------------------------------------------------------------------
// Kernel D: y = relu(x @ W1 + b1), [4096x128]@[128x256]. 2 col-tiles / wave.
// ---------------------------------------------------------------------------
__global__ __launch_bounds__(256) void k_ffn1(const _Float16* __restrict__ x16,
                                              const _Float16* __restrict__ w1t,
                                              const float* __restrict__ b1,
                                              _Float16* __restrict__ y16) {
    const int lane = threadIdx.x & 31;
    const int w    = threadIdx.x >> 5;
    const int rowbase = blockIdx.x * 16;
    const int half = lane >> 4, ll = lane & 15;
    const int j0 = w * 16 + ll, j1 = 128 + w * 16 + ll;

    const _Float16* a0 = x16 + (size_t)rowbase * DD;
    v16h afr[4];
#pragma unroll
    for (int kc = 0; kc < 4; ++kc) afr[kc] = load_a_f16(a0, DD, kc * 32, lane);

    v8f ac0 = {}, ac1 = {};
#pragma unroll
    for (int kc = 0; kc < 4; ++kc) {
        ac0 = wmma_f16(afr[kc], load_b_packed(w1t, DD, j0, kc * 32, lane), ac0);
        ac1 = wmma_f16(afr[kc], load_b_packed(w1t, DD, j1, kc * 32, lane), ac1);
    }

    const float bj0 = b1[j0], bj1 = b1[j1];
#pragma unroll
    for (int r = 0; r < 8; ++r) {
        const size_t grow = (size_t)(rowbase + r + (half << 3));
        y16[grow * (2 * DD) + j0] = (_Float16)fmaxf(ac0[r] + bj0, 0.0f);
        y16[grow * (2 * DD) + j1] = (_Float16)fmaxf(ac1[r] + bj1, 0.0f);
    }
}

// ---------------------------------------------------------------------------
// Kernel E: out = LayerNorm(x + y @ W2 + b2), written fp32 to d_out.
// ---------------------------------------------------------------------------
__global__ __launch_bounds__(256) void k_ffn2_ln2(const _Float16* __restrict__ y16,
                                                  const _Float16* __restrict__ w2t,
                                                  const float* __restrict__ b2,
                                                  const float* __restrict__ x32,
                                                  const float* __restrict__ g2,
                                                  const float* __restrict__ be2,
                                                  float* __restrict__ out) {
    __shared__ float red[16][HH][2];
    const int lane = threadIdx.x & 31;
    const int w    = threadIdx.x >> 5;
    const int rowbase = blockIdx.x * 16;
    const int half = lane >> 4, ll = lane & 15;
    const int j = w * 16 + ll;

    const _Float16* a0 = y16 + (size_t)rowbase * (2 * DD);
    v8f acc = {};
#pragma unroll
    for (int kc = 0; kc < 8; ++kc)
        acc = wmma_f16(load_a_f16(a0, 2 * DD, kc * 32, lane),
                       load_b_packed(w2t, 2 * DD, j, kc * 32, lane), acc);

    const float bj = b2[j];
    float vals[8];
#pragma unroll
    for (int r = 0; r < 8; ++r) {
        const int grow = rowbase + r + (half << 3);
        vals[r] = acc[r] + bj + x32[(size_t)grow * DD + j];
    }

#pragma unroll
    for (int r = 0; r < 8; ++r) {
        float s1 = vals[r], s2 = vals[r] * vals[r];
#pragma unroll
        for (int off = 8; off >= 1; off >>= 1) {
            s1 += __shfl_xor(s1, off, 16);
            s2 += __shfl_xor(s2, off, 16);
        }
        if (ll == 0) {
            const int rr = r + (half << 3);
            red[rr][w][0] = s1;
            red[rr][w][1] = s2;
        }
    }
    __syncthreads();

    const float gj = g2[j], bej = be2[j];
#pragma unroll
    for (int r = 0; r < 8; ++r) {
        const int rr = r + (half << 3);
        float s1 = 0.0f, s2 = 0.0f;
#pragma unroll
        for (int ww = 0; ww < HH; ++ww) { s1 += red[rr][ww][0]; s2 += red[rr][ww][1]; }
        const float mu  = s1 * (1.0f / DD);
        const float var = s2 * (1.0f / DD) - mu * mu;
        const float rs  = rsqrtf(var + 1e-5f);
        out[(size_t)(rowbase + rr) * DD + j] = (vals[r] - mu) * rs * gj + bej;
    }
}

// ---------------------------------------------------------------------------
// Host-side orchestration
// ---------------------------------------------------------------------------
extern "C" void kernel_launch(void* const* d_in, const int* in_sizes, int n_in,
                              void* d_out, int out_size, void* d_ws, size_t ws_size,
                              hipStream_t stream) {
    (void)in_sizes; (void)n_in; (void)out_size; (void)ws_size;

    const float* h     = (const float*)d_in[0];
    const int*   Adj   = (const int*)d_in[1];
    const int*   lens  = (const int*)d_in[2];
    const float* alpha = (const float*)d_in[3];
    const float* Wq = (const float*)d_in[4];  const float* bq = (const float*)d_in[5];
    const float* Wk = (const float*)d_in[6];  const float* bk = (const float*)d_in[7];
    const float* Wv = (const float*)d_in[8];  const float* bv = (const float*)d_in[9];
    const float* Wo = (const float*)d_in[10]; const float* bo = (const float*)d_in[11];
    const float* W1 = (const float*)d_in[12]; const float* b1 = (const float*)d_in[13];
    const float* W2 = (const float*)d_in[14]; const float* b2 = (const float*)d_in[15];
    const float* g1 = (const float*)d_in[16]; const float* be1 = (const float*)d_in[17];
    const float* g2 = (const float*)d_in[18]; const float* be2 = (const float*)d_in[19];
    float* outp = (float*)d_out;

    char* ws = (char*)d_ws;
    size_t off = 0;
    auto carve = [&](size_t bytes) { char* p = ws + off; off += (bytes + 255) & ~(size_t)255; return p; };

    _Float16* qf   = (_Float16*)carve((size_t)BB * HH * NN * DH * 2);   // 1 MB
    _Float16* kf   = (_Float16*)carve((size_t)BB * HH * NN * DH * 2);   // 1 MB
    _Float16* vt   = (_Float16*)carve((size_t)BB * HH * DH * NN * 2);   // 1 MB
    _Float16* attn = (_Float16*)carve((size_t)BB * NN * DD * 2);        // 1 MB
    float*    x32  = (float*)   carve((size_t)BB * NN * DD * 4);        // 2 MB
    _Float16* x16  = (_Float16*)carve((size_t)BB * NN * DD * 2);        // 1 MB
    _Float16* y16  = (_Float16*)carve((size_t)BB * NN * 2 * DD * 2);    // 2 MB
    _Float16* wqt  = (_Float16*)carve((size_t)DD * DD * 2);
    _Float16* wkt  = (_Float16*)carve((size_t)DD * DD * 2);
    _Float16* wvt  = (_Float16*)carve((size_t)DD * DD * 2);
    _Float16* wot  = (_Float16*)carve((size_t)DD * DD * 2);
    _Float16* w1t  = (_Float16*)carve((size_t)DD * 2 * DD * 2);
    _Float16* w2t  = (_Float16*)carve((size_t)2 * DD * DD * 2);

    // pack/transpose weights to f16 B-operand layout
    const int nsq = DD * DD, nrc = DD * 2 * DD;
    k_pack<<<(nsq + 255) / 256, 256, 0, stream>>>(Wq, wqt, DD, DD);
    k_pack<<<(nsq + 255) / 256, 256, 0, stream>>>(Wk, wkt, DD, DD);
    k_pack<<<(nsq + 255) / 256, 256, 0, stream>>>(Wv, wvt, DD, DD);
    k_pack<<<(nsq + 255) / 256, 256, 0, stream>>>(Wo, wot, DD, DD);
    k_pack<<<(nrc + 255) / 256, 256, 0, stream>>>(W1, w1t, DD, 2 * DD);      // -> [256][128]
    k_pack<<<(nrc + 255) / 256, 256, 0, stream>>>(W2, w2t, 2 * DD, DD);      // -> [128][256]

    const int rowTiles = (BB * NN) / 16;  // 256
    k_qkv<<<rowTiles, 256, 0, stream>>>(h, wqt, wkt, wvt, bq, bk, bv, qf, kf, vt);
    k_attn<<<BB * (NN / 16), 256, 0, stream>>>(qf, kf, vt, Adj, lens, alpha, attn);
    k_proj_ln1<<<rowTiles, 256, 0, stream>>>(attn, wot, bo, h, g1, be1, x32, x16);
    k_ffn1<<<rowTiles, 256, 0, stream>>>(x16, w1t, b1, y16);
    k_ffn2_ln2<<<rowTiles, 256, 0, stream>>>(y16, w2t, b2, x32, g2, be2, outp);
}
